// HebbianMambaLayer_89524298318545
// MI455X (gfx1250) — matrix operand order repaired and checked by reading, compiled, not run on hardware
//
#include <hip/hip_runtime.h>
#include <hip/hip_bf16.h>
#include <stdint.h>

// ---------------- problem constants ----------------
#define B_     8
#define L_     2048
#define DM     512          // d_model
#define DI     1024         // d_inner
#define DS     16           // d_state
#define DTR    32           // dt_rank
#define CHUNK_ 64
#define ALPHA_ 0.03f
#define BL     (B_ * L_)    // 16384

// ---------------- WMMA types (gfx1250, wave32) ----------------
typedef __attribute__((ext_vector_type(16))) __bf16 v16bf;
typedef __attribute__((ext_vector_type(8)))  float  v8f;

union FragAB { v16bf v; uint4 q[2]; };
union FragC  { v8f   v; float f[8]; };

__device__ __forceinline__ unsigned short f2bf(float x) {
  unsigned u = __float_as_uint(x);
  unsigned r = u + 0x7FFFu + ((u >> 16) & 1u);   // round-to-nearest-even
  return (unsigned short)(r >> 16);
}

// async-stage one 32x32 bf16 A tile (2 KB) into LDS:
// 128 chunks of 16 B; chunk c = lane + 32*j -> row c>>2, elem offset (c&3)*8
__device__ __forceinline__ void stage_a_async(const unsigned short* Ablk, int lda,
                                              int k0, unsigned ldsoff, int lane)
{
#pragma unroll
  for (int j = 0; j < 4; ++j) {
    int c = lane + 32 * j;
    const unsigned short* g = Ablk + (size_t)(c >> 2) * (size_t)lda + k0 + (c & 3) * 8;
    unsigned l = ldsoff + (unsigned)c * 16u;
    asm volatile("global_load_async_to_lds_b128 %0, %1, off"
                 :: "v"(l), "v"(g) : "memory");
  }
}

// =====================================================================
// Batched bf16 WMMA GEMM:  C[b] = alpha * A[b] @ WT[b]^T + beta * C[b]
// A:  [M,K] bf16 row-major (row stride lda, batch stride sA elements)
// WT: [N,K] bf16 row-major (weight^T; row stride ldw)
// C:  [M,N] f32 (row stride ldc); optional bf16 mirror Cbf (same strides)
// One wave computes a 32x32 tile: 2 A-frags x 2 B-frags -> 4 WMMAs / K-step.
// A tile double-buffered in LDS via global_load_async_to_lds_b128 (ASYNCcnt).
// Requires M,N % 32 == 0, K % 32 == 0 (true for every GEMM in this layer).
// =====================================================================
__global__ void gemm_bf16_wmma(const unsigned short* __restrict__ A, long long sA, int lda,
                               const unsigned short* __restrict__ WT, long long sW, int ldw,
                               float* __restrict__ C, long long sC, int ldc,
                               int M, int N, int K, float alpha, float beta,
                               unsigned short* __restrict__ Cbf)
{
  __shared__ __align__(16) char smem[8 * 2 * 2048];   // 8 waves x double 2 KB buffers
  int wave = (int)(threadIdx.x >> 5);
  int lane = (int)(threadIdx.x & 31);
  int nt2  = (int)blockIdx.x * 8 + wave;              // 32-wide N tile
  int mt2  = (int)blockIdx.y;                         // 32-wide M tile
  if (nt2 * 32 >= N) return;                          // wave-uniform (EXEC all-1 for WMMA)
  int batch = (int)blockIdx.z;
  A  += (size_t)batch * (size_t)sA;
  WT += (size_t)batch * (size_t)sW;
  C  += (size_t)batch * (size_t)sC;
  if (Cbf) Cbf += (size_t)batch * (size_t)sC;

  int row = lane & 15, half = lane >> 4;
  char* wls = smem + wave * 4096;
  unsigned ldsbase = (unsigned)(uintptr_t)wls;        // low 32 bits of LDS aperture addr

  const unsigned short* Ablk  = A + (size_t)mt2 * 32 * (size_t)lda;
  const unsigned short* Brow0 = WT + (size_t)(nt2 * 32 + row) * (size_t)ldw;
  const unsigned short* Brow1 = Brow0 + (size_t)16 * (size_t)ldw;

  FragC acc[2][2];
#pragma unroll
  for (int mi = 0; mi < 2; ++mi)
#pragma unroll
    for (int ni = 0; ni < 2; ++ni)
#pragma unroll
      for (int i = 0; i < 8; ++i) acc[mi][ni].f[i] = 0.f;

  stage_a_async(Ablk, lda, 0, ldsbase, lane);         // prologue -> buf 0
  int buf = 0;

  for (int k0 = 0; k0 < K; k0 += 32) {
    if (k0 + 32 < K) {
      stage_a_async(Ablk, lda, k0 + 32, ldsbase + (unsigned)(buf ^ 1) * 2048u, lane);
      __builtin_prefetch(Brow0 + k0 + 32, 0, 1);      // global_prefetch_b8 next B tile
      __builtin_prefetch(Brow1 + k0 + 32, 0, 1);
      asm volatile("s_wait_asynccnt 4" ::: "memory"); // in-order: current buffer landed
    } else {
      asm volatile("s_wait_asynccnt 0" ::: "memory");
    }

    const char* abase = wls + buf * 2048;
    FragAB a0, a1, b0, b1;
    // A frags from LDS (ds_load_b128): row r at byte r*64; halves at +16*half
    a0.q[0] = *(const uint4*)(abase + row * 64 + 16 * half);
    a0.q[1] = *(const uint4*)(abase + row * 64 + 32 + 16 * half);
    a1.q[0] = *(const uint4*)(abase + (16 + row) * 64 + 16 * half);
    a1.q[1] = *(const uint4*)(abase + (16 + row) * 64 + 32 + 16 * half);
    // B frags direct from global (same K order as A: 8*half and 16+8*half)
    b0.q[0] = *(const uint4*)(Brow0 + k0 + 8 * half);
    b0.q[1] = *(const uint4*)(Brow0 + k0 + 16 + 8 * half);
    b1.q[0] = *(const uint4*)(Brow1 + k0 + 8 * half);
    b1.q[1] = *(const uint4*)(Brow1 + k0 + 16 + 8 * half);

    acc[0][0].v = __builtin_amdgcn_wmma_f32_16x16x32_bf16(false, a0.v, false, b0.v,
                                                          (short)0, acc[0][0].v, false, false);
    acc[0][1].v = __builtin_amdgcn_wmma_f32_16x16x32_bf16(false, a0.v, false, b1.v,
                                                          (short)0, acc[0][1].v, false, false);
    acc[1][0].v = __builtin_amdgcn_wmma_f32_16x16x32_bf16(false, a1.v, false, b0.v,
                                                          (short)0, acc[1][0].v, false, false);
    acc[1][1].v = __builtin_amdgcn_wmma_f32_16x16x32_bf16(false, a1.v, false, b1.v,
                                                          (short)0, acc[1][1].v, false, false);
    buf ^= 1;
  }

#pragma unroll
  for (int mi = 0; mi < 2; ++mi)
#pragma unroll
    for (int ni = 0; ni < 2; ++ni)
#pragma unroll
      for (int i = 0; i < 8; ++i) {
        int m = mt2 * 32 + mi * 16 + i + 8 * half;    // C/D: VGPR i -> row i + 8*half
        int n = nt2 * 32 + ni * 16 + row;
        size_t idx = (size_t)m * (size_t)ldc + (size_t)n;
        float rv = alpha * acc[mi][ni].f[i];
        if (beta != 0.f) rv += beta * C[idx];
        C[idx] = rv;
        if (Cbf) Cbf[idx] = f2bf(rv);
      }
}

// ------------------- transpose + f32->bf16 (batched) -------------------
// out[b][q*P + p] = bf16(in[b][p*Q + q]);  in: [P,Q] f32, out: [Q,P] bf16
__global__ void transpose_to_bf16(const float* __restrict__ in, unsigned short* __restrict__ out,
                                  int P, int Q, long long sIn, long long sOut)
{
  int b = (int)blockIdx.y;
  size_t t = (size_t)blockIdx.x * blockDim.x + threadIdx.x;
  if (t >= (size_t)P * (size_t)Q) return;
  int p = (int)(t % (size_t)P);
  int q = (int)(t / (size_t)P);
  out[(size_t)b * (size_t)sOut + (size_t)q * P + p] =
      f2bf(in[(size_t)b * (size_t)sIn + (size_t)p * Q + q]);
}

// ------------------- RMSNorm (one wave per row of 512) -------------------
__global__ void rmsnorm_bf16(const float* __restrict__ x, const float* __restrict__ w,
                             unsigned short* __restrict__ xnb)
{
  int wave = (int)(threadIdx.x >> 5);
  int lane = (int)(threadIdx.x & 31);
  size_t row = (size_t)blockIdx.x * 8 + wave;
  const float* xr = x + row * DM;
  float vals[16];
  float s = 0.f;
#pragma unroll
  for (int i = 0; i < 16; ++i) { vals[i] = xr[lane + 32 * i]; s += vals[i] * vals[i]; }
#pragma unroll
  for (int off = 16; off >= 1; off >>= 1) s += __shfl_xor(s, off, 32);
  float scale = rsqrtf(s * (1.f / DM) + 1e-5f);
#pragma unroll
  for (int i = 0; i < 16; ++i)
    xnb[row * DM + lane + 32 * i] = f2bf(vals[i] * scale * w[lane + 32 * i]);
}

// --------- causal depthwise conv (D_CONV=4) + bias + SiLU ---------
__global__ void conv_silu(const float* __restrict__ xz, const float* __restrict__ cw,
                          const float* __restrict__ cb,
                          float* __restrict__ u2, unsigned short* __restrict__ u2b)
{
  size_t tid = (size_t)blockIdx.x * blockDim.x + threadIdx.x;
  if (tid >= (size_t)BL * DI) return;
  int c = (int)(tid % DI);
  size_t bl = tid / DI;
  int l = (int)(bl % L_);
  int b = (int)(bl / L_);
  float acc = cb[c];
#pragma unroll
  for (int j = 0; j < 4; ++j) {
    int ls = l - 3 + j;
    if (ls >= 0)
      acc += cw[c * 4 + j] * xz[((size_t)b * L_ + ls) * (2 * DI) + c];
  }
  float s = acc / (1.f + expf(-acc));           // SiLU
  u2[tid] = s;
  u2b[tid] = f2bf(s);
}

// --------- delta = softplus(delta_pre + dt_b[c]) (in place) ---------
__global__ void softplus_bias(float* __restrict__ d, const float* __restrict__ bias, size_t n)
{
  size_t i = (size_t)blockIdx.x * blockDim.x + threadIdx.x;
  if (i >= n) return;
  float v = d[i] + bias[i % DI];
  d[i] = (v > 20.f) ? v : log1pf(expf(v));
}

// --------- selective scan: one thread per (b, channel d), 16 states in regs ---------
__global__ void mamba_scan(const float* __restrict__ delta, const float* __restrict__ u2,
                           const float* __restrict__ xz,   const float* __restrict__ xdbl,
                           const float* __restrict__ A_log, const float* __restrict__ Dw,
                           float* __restrict__ y, unsigned short* __restrict__ yb)
{
  int tid = (int)(blockIdx.x * blockDim.x + threadIdx.x);
  if (tid >= B_ * DI) return;
  int d = tid % DI, b = tid / DI;
  float A[DS], h[DS];
#pragma unroll
  for (int s = 0; s < DS; ++s) { A[s] = -expf(A_log[d * DS + s]); h[s] = 0.f; }
  float Dd = Dw[d];
  for (int l = 0; l < L_; ++l) {
    size_t bl = (size_t)b * L_ + l;
    float dt  = delta[bl * DI + d];
    float ut  = u2[bl * DI + d];
    float res = xz[bl * (2 * DI) + DI + d];
    float du = dt * ut;
    float yv = 0.f;
#pragma unroll
    for (int s = 0; s < DS; ++s) {
      float Bs = xdbl[bl * 64 + DTR + s];
      float Cs = xdbl[bl * 64 + DTR + DS + s];
      h[s] = expf(dt * A[s]) * h[s] + du * Bs;
      yv += h[s] * Cs;
    }
    float o = (yv + ut * Dd) * (res / (1.f + expf(-res)));
    y[bl * DI + d]  = o;
    yb[bl * DI + d] = f2bf(o);
  }
}

__global__ void zero_f32(float* __restrict__ p, size_t n) {
  size_t i = (size_t)blockIdx.x * blockDim.x + threadIdx.x;
  if (i < n) p[i] = 0.f;
}

__global__ void final_add(const float* __restrict__ x, const float* __restrict__ o,
                          const float* __restrict__ r, float* __restrict__ out, size_t n) {
  size_t i = (size_t)blockIdx.x * blockDim.x + threadIdx.x;
  if (i < n) out[i] = x[i] + o[i] + r[i];
}

// =====================================================================
// host side
// =====================================================================
static inline void launch_gemm(const unsigned short* A, long long sA, int lda,
                               const unsigned short* WT, long long sW, int ldw,
                               float* C, long long sC, int ldc,
                               int M, int N, int K, float alpha, float beta,
                               unsigned short* Cbf, int batches, hipStream_t s)
{
  dim3 grid((unsigned)((N / 32 + 7) / 8), (unsigned)(M / 32), (unsigned)batches);
  gemm_bf16_wmma<<<grid, 256, 0, s>>>(A, sA, lda, WT, sW, ldw, C, sC, ldc,
                                      M, N, K, alpha, beta, Cbf);
}

static inline unsigned blocks_for(size_t n) { return (unsigned)((n + 255) / 256); }

extern "C" void kernel_launch(void* const* d_in, const int* in_sizes, int n_in,
                              void* d_out, int out_size, void* d_ws, size_t ws_size,
                              hipStream_t stream) {
  (void)in_sizes; (void)n_in; (void)out_size; (void)ws_size;
  const float* x       = (const float*)d_in[0];
  const float* norm_w  = (const float*)d_in[1];
  const float* in_proj = (const float*)d_in[2];
  const float* conv_w  = (const float*)d_in[3];
  const float* conv_b  = (const float*)d_in[4];
  const float* x_proj  = (const float*)d_in[5];
  const float* dt_w    = (const float*)d_in[6];
  const float* dt_b    = (const float*)d_in[7];
  const float* A_log   = (const float*)d_in[8];
  const float* Dw      = (const float*)d_in[9];
  const float* out_prj = (const float*)d_in[10];
  const float* Wq      = (const float*)d_in[11];
  const float* Wk      = (const float*)d_in[12];
  const float* Wv      = (const float*)d_in[13];
  float* outp = (float*)d_out;

  char* ws = (char*)d_ws;
  size_t off = 0;
  auto alloc = [&](size_t bytes) -> void* {
    void* p = ws + off; off += (bytes + 255) & ~((size_t)255); return p;
  };

  unsigned short* xnb   = (unsigned short*)alloc((size_t)BL * DM * 2);
  unsigned short* wInT  = (unsigned short*)alloc((size_t)(2 * DI) * DM * 2);
  unsigned short* wXpT  = (unsigned short*)alloc((size_t)64 * DI * 2);
  unsigned short* wDtT  = (unsigned short*)alloc((size_t)DI * DTR * 2);
  unsigned short* wOutT = (unsigned short*)alloc((size_t)DM * DI * 2);
  unsigned short* wQT   = (unsigned short*)alloc((size_t)DM * DM * 2);
  unsigned short* wKT   = (unsigned short*)alloc((size_t)DM * DM * 2);
  unsigned short* wVT   = (unsigned short*)alloc((size_t)DM * DM * 2);
  float*          xz    = (float*)alloc((size_t)BL * 2 * DI * 4);
  float*          u2    = (float*)alloc((size_t)BL * DI * 4);
  unsigned short* u2b   = (unsigned short*)alloc((size_t)BL * DI * 2);
  float*          xdbl  = (float*)alloc((size_t)BL * 64 * 4);
  unsigned short* xdblb = (unsigned short*)alloc((size_t)BL * 64 * 2);
  float*          delta = (float*)alloc((size_t)BL * DI * 4);
  float*          yv    = (float*)alloc((size_t)BL * DI * 4);
  unsigned short* yb    = (unsigned short*)alloc((size_t)BL * DI * 2);
  float*          outm  = (float*)alloc((size_t)BL * DM * 4);
  unsigned short* outb  = (unsigned short*)alloc((size_t)BL * DM * 2);
  unsigned short* qb    = (unsigned short*)alloc((size_t)BL * DM * 2);
  float*          tmpf  = (float*)alloc((size_t)BL * DM * 4);
  unsigned short* kTb   = (unsigned short*)alloc((size_t)B_ * DM * L_ * 2);
  unsigned short* vTb   = (unsigned short*)alloc((size_t)B_ * DM * L_ * 2);
  float*          Mst   = (float*)alloc((size_t)B_ * DM * DM * 4);
  unsigned short* MTb   = (unsigned short*)alloc((size_t)B_ * DM * DM * 2);
  float*          rbuf  = (float*)alloc((size_t)BL * DM * 4);

  // ---- weight transpose-converts (W[K,N] f32 -> WT[N,K] bf16) ----
  {
    dim3 g1(blocks_for((size_t)DM * 2 * DI), 1);
    transpose_to_bf16<<<g1, 256, 0, stream>>>(in_proj, wInT, DM, 2 * DI, 0, 0);
    dim3 g2(blocks_for((size_t)DI * 64), 1);
    transpose_to_bf16<<<g2, 256, 0, stream>>>(x_proj, wXpT, DI, 64, 0, 0);
    dim3 g3(blocks_for((size_t)DTR * DI), 1);
    transpose_to_bf16<<<g3, 256, 0, stream>>>(dt_w, wDtT, DTR, DI, 0, 0);
    dim3 g4(blocks_for((size_t)DI * DM), 1);
    transpose_to_bf16<<<g4, 256, 0, stream>>>(out_prj, wOutT, DI, DM, 0, 0);
    dim3 g5(blocks_for((size_t)DM * DM), 1);
    transpose_to_bf16<<<g5, 256, 0, stream>>>(Wq, wQT, DM, DM, 0, 0);
    transpose_to_bf16<<<g5, 256, 0, stream>>>(Wk, wKT, DM, DM, 0, 0);
    transpose_to_bf16<<<g5, 256, 0, stream>>>(Wv, wVT, DM, DM, 0, 0);
  }

  // ---- RMSNorm -> xn bf16 ----
  rmsnorm_bf16<<<BL / 8, 256, 0, stream>>>(x, norm_w, xnb);

  // ---- xz = xn @ in_proj  [BL,2048] ----
  launch_gemm(xnb, 0, DM, wInT, 0, DM, xz, 0, 2 * DI,
              BL, 2 * DI, DM, 1.f, 0.f, nullptr, 1, stream);

  // ---- depthwise conv + SiLU -> u2 (f32 + bf16) ----
  conv_silu<<<blocks_for((size_t)BL * DI), 256, 0, stream>>>(xz, conv_w, conv_b, u2, u2b);

  // ---- x_dbl = u2 @ x_proj  [BL,64] (f32 + bf16 mirror) ----
  launch_gemm(u2b, 0, DI, wXpT, 0, DI, xdbl, 0, 64,
              BL, 64, DI, 1.f, 0.f, xdblb, 1, stream);

  // ---- delta_pre = dt @ dt_w  [BL,1024] (A = x_dbl[:, :32], lda=64) ----
  launch_gemm(xdblb, 0, 64, wDtT, 0, DTR, delta, 0, DI,
              BL, DI, DTR, 1.f, 0.f, nullptr, 1, stream);
  softplus_bias<<<blocks_for((size_t)BL * DI), 256, 0, stream>>>(delta, dt_b, (size_t)BL * DI);

  // ---- selective scan + gate fusion -> y (f32 + bf16) ----
  mamba_scan<<<(B_ * DI) / 256, 256, 0, stream>>>(delta, u2, xz, xdbl, A_log, Dw, yv, yb);

  // ---- out = y @ out_proj  [BL,512] (f32 + bf16 mirror) ----
  launch_gemm(yb, 0, DI, wOutT, 0, DI, outm, 0, DM,
              BL, DM, DI, 1.f, 0.f, outb, 1, stream);

  // ---- q/k/v projections ----
  launch_gemm(outb, 0, DM, wQT, 0, DM, tmpf, 0, DM,
              BL, DM, DM, 1.f, 0.f, qb, 1, stream);          // keep q as bf16
  launch_gemm(outb, 0, DM, wKT, 0, DM, tmpf, 0, DM,
              BL, DM, DM, 1.f, 0.f, nullptr, 1, stream);
  {
    dim3 g(blocks_for((size_t)L_ * DM), (unsigned)B_);       // k -> kT bf16 [B,512,L]
    transpose_to_bf16<<<g, 256, 0, stream>>>(tmpf, kTb, L_, DM,
                                             (long long)L_ * DM, (long long)DM * L_);
  }
  launch_gemm(outb, 0, DM, wVT, 0, DM, tmpf, 0, DM,
              BL, DM, DM, 1.f, 0.f, nullptr, 1, stream);
  {
    dim3 g(blocks_for((size_t)L_ * DM), (unsigned)B_);       // v -> vT bf16 [B,512,L]
    transpose_to_bf16<<<g, 256, 0, stream>>>(tmpf, vTb, L_, DM,
                                             (long long)L_ * DM, (long long)DM * L_);
  }

  // ---- Hebbian chunk scan ----
  zero_f32<<<blocks_for((size_t)B_ * DM * DM), 256, 0, stream>>>(Mst, (size_t)B_ * DM * DM);
  const int S = L_ / CHUNK_;                                  // 32 chunks
  for (int c = 0; c < S; ++c) {
    dim3 gT(blocks_for((size_t)DM * DM), (unsigned)B_);
    transpose_to_bf16<<<gT, 256, 0, stream>>>(Mst, MTb, DM, DM,
                                              (long long)DM * DM, (long long)DM * DM);
    // r_c = q_c @ M   [64,512] per batch (read before update)
    launch_gemm(qb + (size_t)c * CHUNK_ * DM, (long long)L_ * DM, DM,
                MTb, (long long)DM * DM, DM,
                rbuf + (size_t)c * CHUNK_ * DM, (long long)L_ * DM, DM,
                CHUNK_, DM, DM, 1.f, 0.f, nullptr, B_, stream);
    // M = (1-a)*M + (a/64) * k_c^T @ v_c   (K = 64 along chunk axis)
    launch_gemm(kTb + (size_t)c * CHUNK_, (long long)DM * L_, L_,
                vTb + (size_t)c * CHUNK_, (long long)DM * L_, L_,
                Mst, (long long)DM * DM, DM,
                DM, DM, CHUNK_, ALPHA_ / (float)CHUNK_, 1.f - ALPHA_, nullptr, B_, stream);
  }

  // ---- final residual: out = x + out_mamba + r ----
  final_add<<<blocks_for((size_t)BL * DM), 256, 0, stream>>>(x, outm, rbuf, outp,
                                                             (size_t)BL * DM);
}
